// MambaKANPredictor_82523501625844
// MI455X (gfx1250) — matrix operand order repaired and checked by reading, not compiled, not run on hardware
//
#include <hip/hip_runtime.h>
#include <math.h>
#include <stdint.h>

typedef __attribute__((ext_vector_type(16))) _Float16     v16h;
typedef __attribute__((ext_vector_type(8)))  _Float16     v8h;
typedef __attribute__((ext_vector_type(8)))  float        v8f;
typedef __attribute__((ext_vector_type(4)))  unsigned int u32x4;
typedef __attribute__((ext_vector_type(8)))  int          i32x8;
typedef __attribute__((ext_vector_type(4)))  int          i32x4;

#define BSZ    64
#define SEQ    4096
#define IN_DIM 64
#define HDIM   128
#define DSTATE 16
#define HSLD   32              // HS padded row stride (zeros in cols 16..31)
#define MROWS  (BSZ * SEQ)

// ---------------------------------------------------------------------------
// Fast scalar math on CDNA5 transcendental HW
// ---------------------------------------------------------------------------
__device__ __forceinline__ float fast_rcp(float x) {
  return __builtin_amdgcn_rcpf(x);          // v_rcp_f32
}
__device__ __forceinline__ float fast_sigmoid(float x) {
  return fast_rcp(1.f + __expf(-x));        // v_exp_f32 + v_rcp_f32
}
__device__ __forceinline__ float fast_tanh(float x) {
#if __has_builtin(__builtin_amdgcn_tanhf)
  return __builtin_amdgcn_tanhf(x);         // v_tanh_f32
#else
  float r;
  asm("v_tanh_f32 %0, %1" : "=v"(r) : "v"(x));
  return r;
#endif
}
__device__ __forceinline__ float fast_rsqrt(float x) {
#if __has_builtin(__builtin_amdgcn_rsqf)
  return __builtin_amdgcn_rsqf(x);          // v_rsq_f32
#else
  return rsqrtf(x);
#endif
}

// ---------------------------------------------------------------------------
// WMMA helper (CDNA5: D = A(16x32 f16) x B(32x16 f16) + C(16x16 f32))
// ---------------------------------------------------------------------------
__device__ __forceinline__ v8f wmma16(v16h a, v16h b, v8f c) {
  return __builtin_amdgcn_wmma_f32_16x16x32_f16(
      /*neg_a=*/false, a, /*neg_b=*/false, b,
      /*c_mod=*/(short)0, c, /*reuse_a=*/false, /*reuse_b=*/false);
}

// ---------------------------------------------------------------------------
// TDM stage: copy ELEMS f16 (contiguous) from global into LDS.
// D# per CDNA5 ISA 8.3/8.4: group0 = {count=1, lds_addr, global_addr, type=2},
// group1 = {data_size=2B, tensor_dim0=ELEMS, tensor_dim1=1, tile_dim0=ELEMS,
// tile_dim1=1, tensor_dim0_stride=ELEMS}; groups 2/3 zero (<=2D tensor).
// This toolchain's builtin is the 6-arg form: (g0, g1, g2, g3, g4, cpol).
// Issued by wave 0 only; completion via s_wait_tensorcnt + barrier.
// ---------------------------------------------------------------------------
template <int ELEMS>
__device__ __forceinline__ void stage_weights(const _Float16* __restrict__ src,
                                              _Float16* WL, int tid) {
  static_assert(ELEMS <= 65535, "tile_dim0 is 16-bit");
#if __has_builtin(__builtin_amdgcn_tensor_load_to_lds) && \
    __has_builtin(__builtin_amdgcn_s_wait_tensorcnt)
  if (tid < 32) {
    const unsigned lds_off = (unsigned)(uintptr_t)WL;   // flat->LDS: addr[31:0]
    const unsigned long long ga = (unsigned long long)(uintptr_t)src;
    const unsigned n = (unsigned)ELEMS;
    u32x4 g0;
    g0[0] = 1u;                                         // count=1 (user D#)
    g0[1] = lds_off;                                    // lds_addr [63:32]
    g0[2] = (unsigned)ga;                               // global_addr lo
    g0[3] = (unsigned)((ga >> 32) & 0x01ffffffu) | (2u << 30); // ga hi | type=2
    i32x8 g1;
    g1[0] = (int)(1u << 16);                            // data_size=1 (2 bytes)
    g1[1] = (int)((n & 0xffffu) << 16);                 // tensor_dim0 lo16 @bit48
    g1[2] = (int)((n >> 16) & 0xffffu) | (int)(1u << 16); // dim0 hi16 | tensor_dim1=1
    g1[3] = (int)((n & 0xffffu) << 16);                 // tile_dim0 @bit112
    g1[4] = 1;                                          // tile_dim1=1 @bit128
    g1[5] = (int)n;                                     // tensor_dim0_stride lo32
    g1[6] = 0;
    g1[7] = 0;
    i32x4 g2 = {0, 0, 0, 0};
    i32x4 g3 = {0, 0, 0, 0};
    i32x8 g4 = {0, 0, 0, 0, 0, 0, 0, 0};
    __builtin_amdgcn_tensor_load_to_lds(g0, g1, g2, g3, g4, 0);
    __builtin_amdgcn_s_wait_tensorcnt(0);
  }
#else
#pragma unroll 1
  for (int i = tid * 8; i < ELEMS; i += 128 * 8)
    *(v8h*)(WL + i) = *(const v8h*)(src + i);
#endif
  __syncthreads();
}

// A fragment: 16x32 tile of A[M,K] (f32 row-major, stride lda), cvt->f16.
// ISA layout: lane<16 -> row=lane,    K = {kk..kk+7, kk+16..kk+23}
//             lane>=16 -> row=lane-16, K = {kk+8..kk+15, kk+24..kk+31}
__device__ __forceinline__ v16h load_a_frag(const float* __restrict__ A, size_t lda,
                                            int m0, int kk, int lane) {
  const float* row = A + (size_t)(m0 + (lane & 15)) * lda + kk + ((lane < 16) ? 0 : 8);
  const float4 x0 = *(const float4*)(row);
  const float4 x1 = *(const float4*)(row + 4);
  const float4 y0 = *(const float4*)(row + 16);
  const float4 y1 = *(const float4*)(row + 20);
  v16h a;
  a[0]  = (_Float16)x0.x; a[1]  = (_Float16)x0.y;
  a[2]  = (_Float16)x0.z; a[3]  = (_Float16)x0.w;
  a[4]  = (_Float16)x1.x; a[5]  = (_Float16)x1.y;
  a[6]  = (_Float16)x1.z; a[7]  = (_Float16)x1.w;
  a[8]  = (_Float16)y0.x; a[9]  = (_Float16)y0.y;
  a[10] = (_Float16)y0.z; a[11] = (_Float16)y0.w;
  a[12] = (_Float16)y1.x; a[13] = (_Float16)y1.y;
  a[14] = (_Float16)y1.z; a[15] = (_Float16)y1.w;
  return a;
}

// B fragment from LDS-staged f16 weight strip WL[NT*16, LDW] (nrel = n - n0).
// lane<16 -> K={kk..kk+7, kk+16..kk+23} ; lane>=16 -> K={kk+8..kk+15, kk+24..kk+31} (ds_load_b128 x2).
// Padded-K GEMMs overshoot into WL's pad; matching A entries are exact zeros.
template <int LDW>
__device__ __forceinline__ v16h load_b_frag(const _Float16* WL,
                                            int nrel, int kk, int lane) {
  const _Float16* row = WL + nrel * LDW + kk + ((lane < 16) ? 0 : 8);
  const v8h b0 = *(const v8h*)(row);
  const v8h b1 = *(const v8h*)(row + 16);
  v16h b;
#pragma unroll
  for (int j = 0; j < 8; ++j) { b[j] = b0[j]; b[j + 8] = b1[j]; }
  return b;
}

// NT adjacent 16x16 output tiles per wave: one A fragment feeds NT WMMAs.
template <int K, int NT, int LDW>
__device__ __forceinline__ void gemm_tiles(const float* __restrict__ A, size_t lda,
                                           const _Float16* WL,
                                           int m0, int lane, v8f* acc) {
#pragma unroll
  for (int t = 0; t < NT; ++t)
    acc[t] = v8f{0.f, 0.f, 0.f, 0.f, 0.f, 0.f, 0.f, 0.f};
#pragma unroll
  for (int kk = 0; kk < K; kk += 32) {
    v16h a = load_a_frag(A, lda, m0, kk, lane);
#pragma unroll
    for (int t = 0; t < NT; ++t) {
      v16h b = load_b_frag<LDW>(WL, 16 * t, kk, lane);
      acc[t] = wmma16(a, b, acc[t]);
    }
  }
}

// 128-thread workgroup = 4 waves = 4 consecutive M-tiles sharing one LDS
// weight strip of NT*16 columns.
#define GEMM_PRE(K, NT, LDW, Apt, lda, Whp)                                   \
  __shared__ __align__(16) _Float16 WL[(NT) * 16 * (LDW) + 32];               \
  const int tid  = threadIdx.x;                                               \
  const int wave = tid >> 5;                                                  \
  const int lane = tid & 31;                                                  \
  const int m0 = (blockIdx.x * 4 + wave) * 16;                                \
  const int n0 = blockIdx.y * 16 * (NT);                                      \
  const int nl = lane & 15;                                                   \
  const int mbase = m0 + ((lane < 16) ? 0 : 8);                               \
  stage_weights<(NT) * 16 * (LDW)>(Whp + (size_t)n0 * (LDW), WL, tid);        \
  v8f acc[NT];                                                                \
  gemm_tiles<K, NT, LDW>(Apt, lda, WL, m0, lane, acc);

// ---------------------------------------------------------------------------
// f32 -> f16 weight pre-conversion (run once per launch, tiny)
// ---------------------------------------------------------------------------
__global__ __launch_bounds__(256) void cvt_w_kernel(const float* __restrict__ src,
                                                    _Float16* __restrict__ dst, int n) {
  int i = blockIdx.x * 256 + threadIdx.x;
  if (i < n) dst[i] = (_Float16)src[i];
}

// Zero the padded upper halves of HS (cols 16..31 of every 32-wide row).
__global__ __launch_bounds__(256) void init_hs_kernel(float* __restrict__ HS) {
  int i = blockIdx.x * 256 + threadIdx.x;   // i in [0, MROWS*16)
  int m = i >> 4, j = i & 15;
  HS[(size_t)m * HSLD + 16 + j] = 0.f;
}

// ---------------------------------------------------------------------------
// Generic GEMM: C[M,N] = A[M,K] @ Wh[N,K]^T + bias   (bias may be null)
// ---------------------------------------------------------------------------
template <int K, int NT>
__global__ __launch_bounds__(128) void gemm_bias_k(const float* __restrict__ A, size_t lda,
                                                   const _Float16* __restrict__ Wh,
                                                   const float* __restrict__ bias,
                                                   float* __restrict__ C, size_t ldc) {
  GEMM_PRE(K, NT, K, A, lda, Wh);
#pragma unroll
  for (int t = 0; t < NT; ++t) {
    const int n = n0 + 16 * t + nl;
    const float bv = bias ? bias[n] : 0.f;
#pragma unroll
    for (int v = 0; v < 8; ++v) {
      C[(size_t)(mbase + v) * ldc + n] = acc[t][v] + bv;
    }
  }
}

// ---------------------------------------------------------------------------
// y = hs @ Cm^T ; y = (y + x_ssm*Dm) * sigmoid(x_gate) ; in-place into XP[:,n]
// HS is (MROWS x 32) with cols 16..31 zero -> standard K=32 GEMM, LDW=16.
// XP row stride 256: cols [0,128)=x_ssm (overwritten with y), [128,256)=x_gate
// ---------------------------------------------------------------------------
__global__ __launch_bounds__(128) void gemm_gate(const float* __restrict__ HS,
                                                 const _Float16* __restrict__ CmH,
                                                 const float* __restrict__ Dv,
                                                 float* __restrict__ XP) {
  GEMM_PRE(32, 4, DSTATE, HS, HSLD, CmH);
#pragma unroll
  for (int t = 0; t < 4; ++t) {
    const int n = n0 + 16 * t + nl;
    const float dv = Dv[n];
#pragma unroll
    for (int v = 0; v < 8; ++v) {
      size_t r = (size_t)(mbase + v) * (2 * HDIM);
      float xs = XP[r + n];
      float xg = XP[r + HDIM + n];
      XP[r + n] = (acc[t][v] + xs * dv) * fast_sigmoid(xg);
    }
  }
}

// ---------------------------------------------------------------------------
// t = y @ opw^T + opb + residual ; written into XP[:,128+n]
// ---------------------------------------------------------------------------
__global__ __launch_bounds__(128) void gemm_res(const float* __restrict__ XPy, // XP (y in cols 0..127)
                                                const _Float16* __restrict__ WoH,
                                                const float* __restrict__ bo,
                                                const float* __restrict__ Res, // Hbuf, ld 128
                                                float* __restrict__ XP) {
  GEMM_PRE(HDIM, 4, HDIM, XPy, 2 * HDIM, WoH);
#pragma unroll
  for (int t = 0; t < 4; ++t) {
    const int n = n0 + 16 * t + nl;
    const float bv = bo[n];
#pragma unroll
    for (int v = 0; v < 8; ++v) {
      int m = mbase + v;
      XP[(size_t)m * (2 * HDIM) + HDIM + n] = acc[t][v] + bv + Res[(size_t)m * HDIM + n];
    }
  }
}

// ---------------------------------------------------------------------------
// Sequential SSM scan: h_t = tanh(h_{t-1} @ A + xb_t), hs stored to HS
// (padded rows, lower 16 cols only). 1 wave per 2 batches; lane =
// (batch_half, state j); state broadcast via shfl.
// ---------------------------------------------------------------------------
__global__ __launch_bounds__(32) void scan_kernel(const float* __restrict__ XB,
                                                  const float* __restrict__ Amat,
                                                  float* __restrict__ HS) {
  const int lane = threadIdx.x;
  const int j = lane & 15;
  const int b = blockIdx.x * 2 + (lane >> 4);

  float Ac[16];
#pragma unroll
  for (int k = 0; k < 16; ++k) Ac[k] = Amat[k * 16 + j]; // column j of A

  const float* xbp = XB + (size_t)b * SEQ * DSTATE;
  float* hsp = HS + (size_t)b * SEQ * HSLD;

  float h = 0.f;
  float xb_cur = xbp[j];
  for (int t = 0; t < SEQ; ++t) {
    float xb_next = (t + 1 < SEQ) ? xbp[(size_t)(t + 1) * DSTATE + j] : 0.f;
    if (t + 64 < SEQ) __builtin_prefetch(xbp + (size_t)(t + 64) * DSTATE, 0, 0);
    float acc = xb_cur;
#pragma unroll
    for (int k = 0; k < 16; ++k) {
      acc = fmaf(__shfl(h, k, 16), Ac[k], acc); // broadcast h[b,k] in 16-lane group
    }
    h = fast_tanh(acc);                         // single v_tanh_f32 in the chain
    hsp[(size_t)t * HSLD + j] = h;
    xb_cur = xb_next;
  }
}

// ---------------------------------------------------------------------------
// LayerNorm: one wave per row, 4 elems/lane, shfl_xor reductions.
// Input: XP[:,128..256) (ld 256). Output: Hbuf (ld 128).
// ---------------------------------------------------------------------------
__global__ __launch_bounds__(32) void ln_kernel(const float* __restrict__ XP,
                                                const float* __restrict__ g,
                                                const float* __restrict__ bb,
                                                float* __restrict__ Hout) {
  const size_t row = blockIdx.x;
  const int lane = threadIdx.x;
  const float* src = XP + row * (2 * HDIM) + HDIM;
  float v[4], s = 0.f;
#pragma unroll
  for (int i = 0; i < 4; ++i) { v[i] = src[lane + 32 * i]; s += v[i]; }
#pragma unroll
  for (int off = 16; off > 0; off >>= 1) s += __shfl_xor(s, off, 32);
  const float mean = s * (1.f / HDIM);
  float var = 0.f;
#pragma unroll
  for (int i = 0; i < 4; ++i) { float d = v[i] - mean; var += d * d; }
#pragma unroll
  for (int off = 16; off > 0; off >>= 1) var += __shfl_xor(var, off, 32);
  const float inv = fast_rsqrt(var * (1.f / HDIM) + 1e-5f);
#pragma unroll
  for (int i = 0; i < 4; ++i) {
    int c = lane + 32 * i;
    Hout[row * HDIM + c] = (v[i] - mean) * inv * g[c] + bb[c];
  }
}

// ---------------------------------------------------------------------------
// Final head: KAN(last) -> prediction ; MLP(last) -> softplus uncertainty.
// grid = B, block = 64. out[b] = prediction, out[64+b] = uncertainty.
// ---------------------------------------------------------------------------
__global__ __launch_bounds__(64) void head_kernel(
    const float* __restrict__ Hbuf,
    const float* __restrict__ k1bw, const float* __restrict__ k1bb,
    const float* __restrict__ k1sw,
    const float* __restrict__ k2bw, const float* __restrict__ k2bb,
    const float* __restrict__ k2sw,
    const float* __restrict__ u1w, const float* __restrict__ u1b,
    const float* __restrict__ u2w, const float* __restrict__ u2b,
    float* __restrict__ out) {
  const int b = blockIdx.x;
  const int tid = threadIdx.x; // 0..63
  __shared__ float last[HDIM];
  __shared__ float red[64];

  const float* lr = Hbuf + ((size_t)b * SEQ + (SEQ - 1)) * HDIM;
  last[tid] = lr[tid];
  last[tid + 64] = lr[tid + 64];
  __syncthreads();

  const float grid5[5] = {-1.f, -0.5f, 0.f, 0.5f, 1.f};

  // KAN layer 1 (128 -> 64) + uncertainty layer 1 (128 -> 64)
  float base = k1bb[tid];
  float spline = 0.f;
  float ub = u1b[tid];
  for (int h = 0; h < HDIM; ++h) {
    float x = last[h];
    base += x * k1bw[tid * HDIM + h];
    ub += x * u1w[tid * HDIM + h];
    float xc = fminf(1.f, fmaxf(-1.f, x));
#pragma unroll
    for (int gI = 0; gI < 5; ++gI) {
      float d = xc - grid5[gI];
      spline += __expf(-d * d) * k1sw[(size_t)(tid * HDIM + h) * 5 + gI];
    }
  }
  float k1v = fmaxf(0.f, base + spline);
  float uv = fmaxf(0.f, ub);

  // KAN layer 2 (64 -> 1): per-thread contribution, then tree-reduce
  {
    float xc = fminf(1.f, fmaxf(-1.f, k1v));
    float sp = 0.f;
#pragma unroll
    for (int gI = 0; gI < 5; ++gI) {
      float d = xc - grid5[gI];
      sp += __expf(-d * d) * k2sw[tid * 5 + gI];
    }
    red[tid] = k1v * k2bw[tid] + sp;
  }
  __syncthreads();
  for (int s = 32; s > 0; s >>= 1) {
    if (tid < s) red[tid] += red[tid + s];
    __syncthreads();
  }
  if (tid == 0) out[b] = red[0] + k2bb[0];
  __syncthreads();

  // uncertainty layer 2 (64 -> 1) + softplus
  red[tid] = uv * u2w[tid];
  __syncthreads();
  for (int s = 32; s > 0; s >>= 1) {
    if (tid < s) red[tid] += red[tid + s];
    __syncthreads();
  }
  if (tid == 0) {
    float x = red[0] + u2b[0];
    out[BSZ + b] = fmaxf(x, 0.f) + log1pf(__expf(-fabsf(x))); // stable softplus
  }
}

// ---------------------------------------------------------------------------
// Orchestration
// ---------------------------------------------------------------------------
extern "C" void kernel_launch(void* const* d_in, const int* in_sizes, int n_in,
                              void* d_out, int out_size, void* d_ws, size_t ws_size,
                              hipStream_t stream) {
  const float* x        = (const float*)d_in[0];
  const float* emb_w    = (const float*)d_in[1];
  const float* emb_b    = (const float*)d_in[2];
  const float* inproj_w = (const float*)d_in[3];
  const float* inproj_b = (const float*)d_in[4];
  const float* Amat     = (const float*)d_in[5];
  const float* Bm       = (const float*)d_in[6];
  const float* Cm       = (const float*)d_in[7];
  const float* Dm       = (const float*)d_in[8];
  const float* opw      = (const float*)d_in[9];
  const float* opb      = (const float*)d_in[10];
  const float* ln_g     = (const float*)d_in[11];
  const float* ln_b     = (const float*)d_in[12];
  const float* kan1_bw  = (const float*)d_in[13];
  const float* kan1_bb  = (const float*)d_in[14];
  const float* kan1_sw  = (const float*)d_in[15];
  const float* kan2_bw  = (const float*)d_in[16];
  const float* kan2_bb  = (const float*)d_in[17];
  const float* kan2_sw  = (const float*)d_in[18];
  const float* unc1_w   = (const float*)d_in[19];
  const float* unc1_b   = (const float*)d_in[20];
  const float* unc2_w   = (const float*)d_in[21];
  const float* unc2_b   = (const float*)d_in[22];
  float* out = (float*)d_out;

  // workspace: Hbuf | XP | HS(padded) | XB | f16 weights
  float* Hbuf = (float*)d_ws;                        // MROWS * 128
  float* XP   = Hbuf + (size_t)MROWS * HDIM;         // MROWS * 256
  float* HS   = XP + (size_t)MROWS * 2 * HDIM;       // MROWS * 32 (cols 16.. zero)
  float* XB   = HS + (size_t)MROWS * HSLD;           // MROWS * 16

  _Float16* whEmb = (_Float16*)(XB + (size_t)MROWS * DSTATE);
  _Float16* whIp  = whEmb + HDIM * IN_DIM;           // 2 * 256 * 128
  _Float16* whBm  = whIp + 2 * (2 * HDIM) * HDIM;    // 2 * 16 * 128
  _Float16* whCm  = whBm + 2 * DSTATE * HDIM;        // 2 * 128 * 16
  _Float16* whOp  = whCm + 2 * HDIM * DSTATE;        // 2 * 128 * 128

  const dim3 blk(128);
  const int MB = MROWS / 64;   // 4 M-tiles per 128-thread workgroup

  // ---- one-shot setup: weight f16 conversion + HS zero padding ----
  {
    struct { const float* s; _Float16* d; int n; } cv[5] = {
      { emb_w,    whEmb, HDIM * IN_DIM },
      { inproj_w, whIp,  2 * (2 * HDIM) * HDIM },
      { Bm,       whBm,  2 * DSTATE * HDIM },
      { Cm,       whCm,  2 * HDIM * DSTATE },
      { opw,      whOp,  2 * HDIM * HDIM },
    };
    for (int c = 0; c < 5; ++c)
      cvt_w_kernel<<<dim3((cv[c].n + 255) / 256), dim3(256), 0, stream>>>(cv[c].s, cv[c].d, cv[c].n);
    init_hs_kernel<<<dim3((MROWS * 16) / 256), dim3(256), 0, stream>>>(HS);
  }

  // h = x @ emb_w^T + emb_b  (K=64, N=128 -> 2 strips of 64)
  gemm_bias_k<IN_DIM, 4><<<dim3(MB, 2), blk, 0, stream>>>(x, IN_DIM, whEmb, emb_b, Hbuf, HDIM);

  for (int i = 0; i < 2; ++i) {
    const _Float16* ipwH = whIp + (size_t)i * 2 * HDIM * HDIM;
    const _Float16* BmH  = whBm + (size_t)i * DSTATE * HDIM;
    const _Float16* CmH  = whCm + (size_t)i * HDIM * DSTATE;
    const _Float16* WoH  = whOp + (size_t)i * HDIM * HDIM;
    const float* ipb = inproj_b + (size_t)i * 2 * HDIM;
    const float* Ai  = Amat + (size_t)i * DSTATE * DSTATE;
    const float* Dmi = Dm + (size_t)i * HDIM;
    const float* boi = opb + (size_t)i * HDIM;
    const float* gi  = ln_g + (size_t)i * HDIM;
    const float* bi  = ln_b + (size_t)i * HDIM;

    // xp = h @ ipw^T + ipb  -> XP (cols 0..127 = x_ssm, 128..255 = x_gate)
    gemm_bias_k<HDIM, 4><<<dim3(MB, 4), blk, 0, stream>>>(Hbuf, HDIM, ipwH, ipb, XP, 2 * HDIM);
    // xb = x_ssm @ Bm^T -> XB  (K=128, N=16)
    gemm_bias_k<HDIM, 1><<<dim3(MB, 1), blk, 0, stream>>>(XP, 2 * HDIM, BmH, nullptr, XB, DSTATE);
    // sequential scan -> HS (lower 16 cols of padded rows)
    scan_kernel<<<dim3(BSZ / 2), dim3(32), 0, stream>>>(XB, Ai, HS);
    // y = (hs@Cm^T + x_ssm*Dm) * sigmoid(x_gate) -> XP cols 0..127 (in-place)
    gemm_gate<<<dim3(MB, 2), blk, 0, stream>>>(HS, CmH, Dmi, XP);
    // t = y@opw^T + opb + residual -> XP cols 128..255
    gemm_res<<<dim3(MB, 2), blk, 0, stream>>>(XP, WoH, boi, Hbuf, XP);
    // h = LN(t) -> Hbuf
    ln_kernel<<<dim3(MROWS), dim3(32), 0, stream>>>(XP, gi, bi, Hbuf);
  }

  head_kernel<<<dim3(BSZ), dim3(64), 0, stream>>>(
      Hbuf, kan1_bw, kan1_bb, kan1_sw, kan2_bw, kan2_bb, kan2_sw,
      unc1_w, unc1_b, unc2_w, unc2_b, out);
}
